// RPScaledDPAttention_14491219656893
// MI455X (gfx1250) — matrix operand order repaired and checked by reading, not compiled, not run on hardware
//
#include <hip/hip_runtime.h>

typedef __attribute__((ext_vector_type(16))) __bf16 bf16x16;
typedef __attribute__((ext_vector_type(8)))  __bf16 bf16x8;
typedef __attribute__((ext_vector_type(2)))  __bf16 bf16x2;
typedef __attribute__((ext_vector_type(8)))  float  f32x8;
typedef __attribute__((ext_vector_type(4)))  float  f32x4;

#define NH    16
#define SEQ   1024
#define DHEAD 64
#define NPOS  301
#define QPS   305   /* qp row stride (f32), co-prime with 64 banks -> conflict-free rows */
#define KSTR  72    /* K tile row stride (halfs), 144B, 16B aligned */
#define VSTR  40    /* V^T tile row stride (halfs), 80B, 16B aligned */
#define BQ    64
#define KBLK  32
#define NEG_BIG (-3.0e38f)

#define WMMA_BF16(A, B, C) \
  __builtin_amdgcn_wmma_f32_16x16x32_bf16(false, (A), false, (B), (short)0, (C), false, false)

__device__ inline f32x8 zero8() {
  f32x8 z;
  #pragma unroll
  for (int i = 0; i < 8; ++i) z[i] = 0.0f;
  return z;
}

// A operand (16x32, 16-bit), this file's contiguous k order: element j = M[m][hi*16 + j]
__device__ inline bf16x16 frag_A_f32(const float* row, int hi, float s) {
  const f32x4* pa = (const f32x4*)(row + hi * 16);
  const f32x4* pb = (const f32x4*)(row + hi * 16 + 8);
  f32x4 x0 = pa[0], x1 = pa[1], y0 = pb[0], y1 = pb[1];
  bf16x16 r;
  #pragma unroll
  for (int i = 0; i < 4; ++i) {
    r[i]      = (__bf16)(x0[i] * s);
    r[4 + i]  = (__bf16)(x1[i] * s);
    r[8 + i]  = (__bf16)(y0[i] * s);
    r[12 + i] = (__bf16)(y1[i] * s);
  }
  return r;
}

// B operand (32x16, 16-bit), this file's contiguous k order: element j = B[hi*16 + j][n]; for B = X^T this is X[row][hi*16+j]
__device__ inline bf16x16 frag_B_f32(const float* row, int hi, float s) {
  const f32x4* p = (const f32x4*)(row + hi * 16);
  f32x4 x0 = p[0], x1 = p[1], x2 = p[2], x3 = p[3];
  bf16x16 r;
  #pragma unroll
  for (int i = 0; i < 4; ++i) {
    r[i]      = (__bf16)(x0[i] * s);
    r[4 + i]  = (__bf16)(x1[i] * s);
    r[8 + i]  = (__bf16)(x2[i] * s);
    r[12 + i] = (__bf16)(x3[i] * s);
  }
  return r;
}

// A-layout fragment from a row-major bf16 LDS row (two 16B ds loads)
__device__ inline bf16x16 frag_A_lds(const __bf16* row, int hi) {
  const bf16x8* p = (const bf16x8*)(row + hi * 16);
  bf16x8 a = p[0];   // K = hi*16 .. +7
  bf16x8 b = p[1];   // K = hi*16+8 .. +15
  return __builtin_shufflevector(a, b, 0,1,2,3,4,5,6,7,8,9,10,11,12,13,14,15);
}

// issue global loads for one K/V tile into registers (2 keys x 8 dims per thread)
__device__ inline void tile_issue(const float* kbg, const float* vbg,
                                  int kb0, int kk0, int dd,
                                  f32x4 kr[4], f32x4 vr[4]) {
  const f32x4* k0 = (const f32x4*)(kbg + (long)(kb0 + kk0)     * DHEAD + dd);
  const f32x4* k1 = (const f32x4*)(kbg + (long)(kb0 + kk0 + 1) * DHEAD + dd);
  const f32x4* v0 = (const f32x4*)(vbg + (long)(kb0 + kk0)     * DHEAD + dd);
  const f32x4* v1 = (const f32x4*)(vbg + (long)(kb0 + kk0 + 1) * DHEAD + dd);
  kr[0] = k0[0]; kr[1] = k0[1]; kr[2] = k1[0]; kr[3] = k1[1];
  vr[0] = v0[0]; vr[1] = v0[1]; vr[2] = v1[0]; vr[3] = v1[1];
}

// convert + store registers into the bf16 K tile and transposed V tile
__device__ inline void tile_store(__bf16* kt, __bf16* vt, int kk0, int dd,
                                  const f32x4 kr[4], const f32x4 vr[4]) {
  bf16x8 a, b;
  #pragma unroll
  for (int i = 0; i < 4; ++i) {
    a[i] = (__bf16)kr[0][i]; a[4 + i] = (__bf16)kr[1][i];
    b[i] = (__bf16)kr[2][i]; b[4 + i] = (__bf16)kr[3][i];
  }
  *(bf16x8*)(kt + kk0 * KSTR + dd)       = a;   // 16B aligned ds_store_b128
  *(bf16x8*)(kt + (kk0 + 1) * KSTR + dd) = b;
  #pragma unroll
  for (int i = 0; i < 4; ++i) {
    bf16x2 p0; p0[0] = (__bf16)vr[0][i]; p0[1] = (__bf16)vr[2][i];
    bf16x2 p1; p1[0] = (__bf16)vr[1][i]; p1[1] = (__bf16)vr[3][i];
    *(bf16x2*)(vt + (dd + i)     * VSTR + kk0) = p0;   // packed 4B transpose stores
    *(bf16x2*)(vt + (dd + 4 + i) * VSTR + kk0) = p1;
  }
}

__global__ __launch_bounds__(128) void rp_attn_kernel(
    const float* __restrict__ gq, const float* __restrict__ gk,
    const float* __restrict__ gv, const float* __restrict__ gpe,
    const unsigned char* __restrict__ gmask, float* __restrict__ gout)
{
  __shared__ __align__(16) float   qp[BQ * QPS];        // 78,080 B
  __shared__ __align__(16) __bf16  kt[KBLK * KSTR];     //  4,608 B
  __shared__ __align__(16) __bf16  vt[DHEAD * VSTR];    //  5,120 B
  __shared__ __align__(8)  unsigned char smask[SEQ];    //  1,024 B

  const int tid  = threadIdx.x;
  const int wave = tid >> 5;
  const int lane = tid & 31;
  const int lo   = lane & 15;
  const int hi   = lane >> 4;

  const int bh   = blockIdx.x >> 4;       // 64 (b,h) pairs
  const int qblk = blockIdx.x & 15;       // 16 query blocks of 64
  const int b    = bh >> 4;               // H = 16
  const long baseBH = (long)bh * SEQ * DHEAD;
  const int qbase = qblk * BQ + wave * 16;

  // stage this batch's key mask row into LDS (128 threads x 8B = 1024B)
  ((unsigned long long*)smask)[tid] =
      ((const unsigned long long*)(gmask + (long)b * SEQ))[tid];

  const float* kbg = gk + baseBH;
  const float* vbg = gv + baseBH;
  const int kk0 = (tid & 15) * 2;     // 2 adjacent keys per thread
  const int ddl = (tid >> 4) * 8;     // 8 dims per thread

  // prefetch tile 0 into registers; completes while the QP GEMM runs
  f32x4 krg[4], vrg[4];
  tile_issue(kbg, vbg, 0, kk0, ddl, krg, vrg);

  const float* qrow = gq + baseBH + (long)(qbase + lo) * DHEAD;

  // ---- Phase A: QP = (Q * 1/8) @ PosEmb^T -> [16 rows x 304] f32 per wave ----
  {
    bf16x16 qa0 = frag_A_f32(qrow,      hi, 0.125f);
    bf16x16 qa1 = frag_A_f32(qrow + 32, hi, 0.125f);
    for (int t = 0; t < 19; ++t) {
      const int r = t * 16 + lo;
      bf16x16 pe0, pe1;
      if (r < NPOS) {
        const float* prow = gpe + r * DHEAD;
        pe0 = frag_B_f32(prow,      hi, 1.0f);
        pe1 = frag_B_f32(prow + 32, hi, 1.0f);
      } else {
        #pragma unroll
        for (int i = 0; i < 16; ++i) { pe0[i] = (__bf16)0.0f; pe1[i] = (__bf16)0.0f; }
      }
      f32x8 c = zero8();
      c = WMMA_BF16(qa0, pe0, c);
      c = WMMA_BF16(qa1, pe1, c);
      const int rowb = wave * 16 + 8 * hi;   // C layout: M = vv + 8*hi, N = lo
      #pragma unroll
      for (int vv = 0; vv < 8; ++vv)
        qp[(rowb + vv) * QPS + t * 16 + lo] = c[vv];
    }
  }

  // persistent Q^T B-layout fragments (reused every key block)
  const bf16x16 qb0 = frag_B_f32(qrow,      hi, 0.125f);
  const bf16x16 qb1 = frag_B_f32(qrow + 32, hi, 0.125f);

  f32x8 oacc[4];
  #pragma unroll
  for (int f = 0; f < 4; ++f) oacc[f] = zero8();
  float mrow = NEG_BIG, lrow = 0.0f;

  const int    qg    = qbase + lo;                      // this lane's global query row
  const float* qprow = qp + (wave * 16 + lo) * QPS;

  for (int kb0 = 0; kb0 < SEQ; kb0 += KBLK) {
    __syncthreads();                       // previous tile fully consumed
    tile_store(kt, vt, kk0, ddl, krg, vrg);
    __syncthreads();                       // tile ready
    if (kb0 + KBLK < SEQ)                  // uniform branch: prefetch next tile
      tile_issue(kbg, vbg, kb0 + KBLK, kk0, ddl, krg, vrg);

    // S^T = K @ Q^T : lane = query, VGPR = key
    f32x8 sarr[2];
    #pragma unroll
    for (int g = 0; g < 2; ++g) {
      const __bf16* krow = kt + (g * 16 + lo) * KSTR;
      bf16x16 ka0 = frag_A_lds(krow,      hi);
      bf16x16 ka1 = frag_A_lds(krow + 32, hi);
      f32x8 c = zero8();
      c = WMMA_BF16(ka0, qb0, c);
      c = WMMA_BF16(ka1, qb1, c);
      sarr[g] = c;
    }

    // add relative-position term (QP gather) + mask; running row max
    float mloc = NEG_BIG;
    #pragma unroll
    for (int g = 0; g < 2; ++g) {
      #pragma unroll
      for (int vv = 0; vv < 8; ++vv) {
        const int key = kb0 + g * 16 + 8 * hi + vv;
        int rel = key - qg;
        int idx = rel < -150 ? -150 : (rel > 150 ? 150 : rel);
        idx += 149;
        if (idx < 0) idx = 300;            // reference's -1 wrap to last row
        float sc = sarr[g][vv] + qprow[idx];
        sc = smask[key] ? NEG_BIG : sc;
        sarr[g][vv] = sc;
        mloc = fmaxf(mloc, sc);
      }
    }
    const float mnew  = fmaxf(mrow, fmaxf(mloc, __shfl_xor(mloc, 16, 32)));
    const float alpha = __expf(mrow - mnew);
    mrow = mnew;

    float p0v[8], p1v[8];
    float psum = 0.0f;
    #pragma unroll
    for (int vv = 0; vv < 8; ++vv) {
      p0v[vv] = __expf(sarr[0][vv] - mnew);
      p1v[vv] = __expf(sarr[1][vv] - mnew);
      psum += p0v[vv] + p1v[vv];
    }
    psum += __shfl_xor(psum, 16, 32);
    lrow = alpha * lrow + psum;

    // build P^T B-fragment: lane needs keys hi*16 .. hi*16+15 -> one xor-16 exchange
    bf16x16 pb;
    #pragma unroll
    for (int j = 0; j < 8; ++j) {
      const float send = hi ? p0v[j] : p1v[j];
      const float recv = __shfl_xor(send, 16, 32);
      pb[j]     = (__bf16)(hi ? recv    : p0v[j]);
      pb[8 + j] = (__bf16)(hi ? p1v[j] : recv);
    }

    // O^T += V^T @ P^T (4 dim tiles), rescaled by online-softmax alpha
    #pragma unroll
    for (int f = 0; f < 4; ++f) {
      #pragma unroll
      for (int vv = 0; vv < 8; ++vv) oacc[f][vv] *= alpha;
      bf16x16 va = frag_A_lds(vt + (f * 16 + lo) * VSTR, hi);
      oacc[f] = WMMA_BF16(va, pb, oacc[f]);
    }
  }

  // epilogue: O[q][d] = O^T C-layout / l ; d = f*16 + 8*hi + vv (8 consecutive floats)
  const float inv = 1.0f / lrow;
  float* orow = gout + baseBH + (long)(qbase + lo) * DHEAD;
  #pragma unroll
  for (int f = 0; f < 4; ++f) {
    const int dd = f * 16 + 8 * hi;
    f32x4 w0, w1;
    #pragma unroll
    for (int i = 0; i < 4; ++i) { w0[i] = oacc[f][i] * inv; w1[i] = oacc[f][4 + i] * inv; }
    *(f32x4*)(orow + dd)     = w0;
    *(f32x4*)(orow + dd + 4) = w1;
  }
}

extern "C" void kernel_launch(void* const* d_in, const int* in_sizes, int n_in,
                              void* d_out, int out_size, void* d_ws, size_t ws_size,
                              hipStream_t stream) {
  const float* gq  = (const float*)d_in[0];
  const float* gk  = (const float*)d_in[1];
  const float* gv  = (const float*)d_in[2];
  const float* gpe = (const float*)d_in[3];
  const unsigned char* gmask = (const unsigned char*)d_in[4];  // jax bool = 1 byte
  float* gout = (float*)d_out;

  const int nblocks = 4 * NH * (SEQ / BQ);  // 1024 workgroups
  rp_attn_kernel<<<nblocks, 128, 0, stream>>>(gq, gk, gv, gpe, gmask, gout);

  (void)in_sizes; (void)n_in; (void)out_size; (void)d_ws; (void)ws_size;
}
